// pcn_11441792876785
// MI455X (gfx1250) — matrix operand order repaired and checked by reading, not compiled, not run on hardware
//
#include <hip/hip_runtime.h>
#include <hip/hip_bf16.h>
#include <cstddef>

// ---------------------------------------------------------------------------
// MI455X / gfx1250 (CDNA5, wave32). All heavy GEMMs use v_wmma_f32_16x16x32_f16
// (f16 A/B, f32 accumulate). A/B fragments use contiguous k = 16*(lane/16)+s
// in both A and B (not the ISA table k order).
// ---------------------------------------------------------------------------

typedef _Float16 half_t;
typedef __attribute__((ext_vector_type(16))) _Float16 v16h;
typedef __attribute__((ext_vector_type(8)))  float    v8f;

#define WMMA_F16(a, b, c) \
  __builtin_amdgcn_wmma_f32_16x16x32_f16(false, (a), false, (b), (short)0, (c), false, false)

// ---- fragment builders (contiguous-k 16-bit layouts, wave32) ----------------
// A (16x32, f16): lane<16: row=lane, K0..15
//                 lane>=16: row=lane-16, K16..31
__device__ __forceinline__ v16h load_a_frag_lds(const half_t* As, int ldh,
                                                int rowbase, int lane) {
  const int r  = rowbase + (lane & 15);
  const int kb = (lane < 16) ? 0 : 16;
  const half_t* row = As + (size_t)r * ldh;
  v16h a;
#pragma unroll
  for (int i = 0; i < 8; ++i) a[i] = row[kb + i];
#pragma unroll
  for (int i = 0; i < 8; ++i) a[8 + i] = row[kb + 8 + i];
  return a;
}

// B (32x16, f16): lane<16: col=lane, K0..15; lane>=16: col=lane-16, K16..31
__device__ __forceinline__ v16h load_b_frag_lds(const half_t* Ws, int ldh,
                                                int nbase, int lane) {
  const int n    = nbase + (lane & 15);
  const int koff = (lane >> 4) * 16;
  const half_t* row = Ws + (size_t)n * ldh + koff;
  v16h b;
#pragma unroll
  for (int i = 0; i < 16; ++i) b[i] = row[i];
  return b;
}

// ---------------------------------------------------------------------------
// Generic GEMM: C(M,N) = act(A(M,K) @ W(N,K)^T + bias), all f32 in memory,
// f16 WMMA inside.  MT=64, NT=128, KT=32, 256 threads = 8 waves (2m x 4n),
// each wave owns a 32x32 C tile (4 accum frags).
// Requires K % 32 == 0; M,N guarded.
// ---------------------------------------------------------------------------
template <bool RELU>
__global__ __launch_bounds__(256) void gemm_wmma(const float* __restrict__ A,
                                                 const float* __restrict__ W,
                                                 const float* __restrict__ bias,
                                                 float* __restrict__ C,
                                                 int M, int N, int K) {
  constexpr int MT = 64, NT = 128, KT = 32, LDH = KT + 4;
  __shared__ half_t As[MT][LDH];
  __shared__ half_t Ws[NT][LDH];

  const int tid  = threadIdx.x;
  const int lane = tid & 31;
  const int wv   = tid >> 5;
  const int wm   = wv & 1;   // 0..1 -> m offset wm*32
  const int wn   = wv >> 1;  // 0..3 -> n offset wn*32
  const int mbase = blockIdx.x * MT;
  const int nbase = blockIdx.y * NT;

  v8f acc[2][2] = {};

  for (int k0 = 0; k0 < K; k0 += KT) {
    __syncthreads();
    {  // stage A tile: 64x32 f32 -> f16, 8 elems/thread (two float4)
      const int r  = tid >> 2;
      const int kk = (tid & 3) * 8;
      const int gm = mbase + r;
      float4 v0, v1;
      if (gm < M) {
        const float* src = A + (size_t)gm * K + k0 + kk;
        v0 = *(const float4*)(src);
        v1 = *(const float4*)(src + 4);
      } else {
        v0 = make_float4(0.f, 0.f, 0.f, 0.f);
        v1 = v0;
      }
      half_t* d = &As[r][kk];
      d[0] = (half_t)v0.x; d[1] = (half_t)v0.y; d[2] = (half_t)v0.z; d[3] = (half_t)v0.w;
      d[4] = (half_t)v1.x; d[5] = (half_t)v1.y; d[6] = (half_t)v1.z; d[7] = (half_t)v1.w;
    }
    {  // stage W tile: 128x32 f32 -> f16, 16 elems/thread
      const int r  = tid >> 1;
      const int kk = (tid & 1) * 16;
      const int gn = nbase + r;
      float4 v[4];
      if (gn < N) {
        const float* src = W + (size_t)gn * K + k0 + kk;
#pragma unroll
        for (int q = 0; q < 4; ++q) v[q] = *(const float4*)(src + q * 4);
      } else {
#pragma unroll
        for (int q = 0; q < 4; ++q) v[q] = make_float4(0.f, 0.f, 0.f, 0.f);
      }
      half_t* d = &Ws[r][kk];
#pragma unroll
      for (int q = 0; q < 4; ++q) {
        d[q * 4 + 0] = (half_t)v[q].x; d[q * 4 + 1] = (half_t)v[q].y;
        d[q * 4 + 2] = (half_t)v[q].z; d[q * 4 + 3] = (half_t)v[q].w;
      }
    }
    __syncthreads();

    const v16h a0 = load_a_frag_lds(&As[0][0], LDH, wm * 32, lane);
    const v16h a1 = load_a_frag_lds(&As[0][0], LDH, wm * 32 + 16, lane);
#pragma unroll
    for (int an = 0; an < 2; ++an) {
      const v16h bf = load_b_frag_lds(&Ws[0][0], LDH, wn * 32 + an * 16, lane);
      acc[0][an] = WMMA_F16(a0, bf, acc[0][an]);
      acc[1][an] = WMMA_F16(a1, bf, acc[1][an]);
    }
  }

  // store: C frag layout — VGPR r holds M=r (lane<16) or M=8+r (lane>=16)
  const int mrow = (lane >> 4) * 8;
  const int ncol = lane & 15;
#pragma unroll
  for (int am = 0; am < 2; ++am)
#pragma unroll
    for (int an = 0; an < 2; ++an)
#pragma unroll
      for (int r = 0; r < 8; ++r) {
        const int m = mbase + wm * 32 + am * 16 + mrow + r;
        const int n = nbase + wn * 32 + an * 16 + ncol;
        if (m < M && n < N) {
          float v = acc[am][an][r] + bias[n];
          if (RELU) v = fmaxf(v, 0.f);
          C[(size_t)m * N + n] = v;
        }
      }
}

// ---------------------------------------------------------------------------
// enc layer 1: out(16384,128) = relu(pts @ enc_w1^T + b1), K=3 -> scalar
// x layout (B,3,N)
// ---------------------------------------------------------------------------
__global__ void enc1_kernel(const float* __restrict__ x,
                            const float* __restrict__ w1,
                            const float* __restrict__ b1,
                            float* __restrict__ out) {
  const int idx = blockIdx.x * blockDim.x + threadIdx.x;
  if (idx >= 16384 * 128) return;
  const int r = idx >> 7, o = idx & 127;
  const int b = r >> 11, n = r & 2047;
  const float* xb = x + (size_t)b * 3 * 2048;
  const float v = b1[o] + w1[o * 3 + 0] * xb[n] + w1[o * 3 + 1] * xb[2048 + n] +
                  w1[o * 3 + 2] * xb[4096 + n];
  out[idx] = fmaxf(v, 0.f);
}

// out[b*C+j] = max over 2048 points of in[(b*2048+n)*C + j]
__global__ void rowmax_kernel(const float* __restrict__ in,
                              float* __restrict__ out, int C) {
  const int idx = blockIdx.x * blockDim.x + threadIdx.x;
  if (idx >= 8 * C) return;
  const int b = idx / C, j = idx % C;
  const float* p = in + (size_t)b * 2048 * C + j;
  float m = -3.0e38f;
  for (int n = 0; n < 2048; ++n) m = fmaxf(m, p[(size_t)n * C]);
  out[idx] = m;
}

// cat(16384,512): [0:256)=h row, [256:512)=g[batch]
__global__ void concat_kernel(const float* __restrict__ h,
                              const float* __restrict__ g,
                              float* __restrict__ out) {
  const int idx = blockIdx.x * blockDim.x + threadIdx.x;
  if (idx >= 16384 * 512) return;
  const int r = idx >> 9, j = idx & 511;
  const int b = r >> 11;
  out[idx] = (j < 256) ? h[(size_t)r * 256 + j] : g[b * 256 + (j - 256)];
}

// base(8,512): t_glob + f2_b1  (K=1024 scalar dot, tiny)
__global__ void base_kernel(const float* __restrict__ feature,
                            const float* __restrict__ f2_w1,
                            const float* __restrict__ f2_b1,
                            float* __restrict__ out) {
  const int idx = blockIdx.x * blockDim.x + threadIdx.x;
  if (idx >= 8 * 512) return;
  const int b = idx >> 9, k = idx & 511;
  const float* wr = f2_w1 + (size_t)k * 1029 + 5;
  const float* fb = feature + b * 1024;
  float s = f2_b1[k];
  for (int j = 0; j < 1024; ++j) s += fb[j] * wr[j];
  out[idx] = s;
}

// t_grid(16,512) from linspace(-0.05,0.05,4) meshgrid('ij')
__global__ void tgrid_kernel(const float* __restrict__ f2_w1,
                             float* __restrict__ out) {
  const int idx = blockIdx.x * blockDim.x + threadIdx.x;
  if (idx >= 16 * 512) return;
  const int g = idx >> 9, k = idx & 511;
  const float step = 0.1f / 3.f;
  const float gx = -0.05f + step * (float)(g >> 2);
  const float gy = -0.05f + step * (float)(g & 3);
  const float* wr = f2_w1 + (size_t)k * 1029;
  out[idx] = gx * wr[0] + gy * wr[1];
}

// ---------------------------------------------------------------------------
// Fused folding-2: for each WG: 32 h1 rows (2 coarse pts x 16 grid) x N=512.
//   h1 = relu(t_grid[g] + t_pt[b,p] + (t_glob[b]+f2_b1))        (built in LDS)
//   h2 = relu(h1 @ f2_w2^T + f2_b2)                             (WMMA, K=512)
//   fine = h2 @ f2_w3^T + f2_b3 + coarse[b,p]                   (N=3 tail)
// Never materializes h1/h2 in HBM. B-frags read f2_w2 (1MB, L2-hot) directly.
// ---------------------------------------------------------------------------
__global__ __launch_bounds__(256) void fold2_fused(
    const float* __restrict__ coarse,  // (8,1024,3) == d_out[0:24576]
    const float* __restrict__ baseb,   // (8,512)  t_glob + f2_b1
    const float* __restrict__ tgrid,   // (16,512)
    const float* __restrict__ f2_w1,   // (512,1029)  cols 2..4 = W_pt
    const float* __restrict__ f2_w2,   // (512,512)
    const float* __restrict__ f2_b2,   // (512,)
    const float* __restrict__ f2_w3,   // (3,512)
    const float* __restrict__ f2_b3,   // (3,)
    float* __restrict__ fine) {        // (8,16384,3)
  constexpr int KTOT = 512, KT = 32, LDH = KT + 4;
  __shared__ half_t As[32][LDH];
  __shared__ float  comb[2][KT];
  __shared__ half_t Hs[32][KTOT + 8];  // relu(h2) staged for the N=3 tail

  const int tid  = threadIdx.x;
  const int lane = tid & 31;
  const int wv   = tid >> 5;                // wave owns n in [wv*64, wv*64+64)
  const int tile = blockIdx.x;              // 4096 tiles
  const int b    = tile >> 9;               // 512 tiles per batch
  const int p0   = (tile & 511) * 2;        // two coarse points per tile

  v8f acc[2][4] = {};

  for (int k0 = 0; k0 < KTOT; k0 += KT) {
    __syncthreads();
    if (tid < 64) {  // comb[slot][k] = base + t_pt (3 MACs from coarse)
      const int slot = tid >> 5, k = tid & 31, gk = k0 + k;
      const float* cr = coarse + (size_t)b * 3072 + (p0 + slot) * 3;
      const float* wr = f2_w1 + (size_t)gk * 1029;
      comb[slot][k] = baseb[b * 512 + gk] + cr[0] * wr[2] + cr[1] * wr[3] + cr[2] * wr[4];
    }
    __syncthreads();
    {  // build h1 tile (32x32) in f16: add t_grid, relu
      const int r = tid >> 3;
      const int k = (tid & 7) * 4;
      const int g = r & 15, slot = r >> 4;
      const float4 t = *(const float4*)(tgrid + (size_t)g * 512 + k0 + k);
      As[r][k + 0] = (half_t)fmaxf(t.x + comb[slot][k + 0], 0.f);
      As[r][k + 1] = (half_t)fmaxf(t.y + comb[slot][k + 1], 0.f);
      As[r][k + 2] = (half_t)fmaxf(t.z + comb[slot][k + 2], 0.f);
      As[r][k + 3] = (half_t)fmaxf(t.w + comb[slot][k + 3], 0.f);
    }
    __syncthreads();

    const v16h a0 = load_a_frag_lds(&As[0][0], LDH, 0, lane);
    const v16h a1 = load_a_frag_lds(&As[0][0], LDH, 16, lane);
    const int koff = (lane >> 4) * 16;
#pragma unroll
    for (int an = 0; an < 4; ++an) {
      const int n = wv * 64 + an * 16 + (lane & 15);
      const float* wrow = f2_w2 + (size_t)n * 512 + k0 + koff;
      v16h bf;
#pragma unroll
      for (int i = 0; i < 16; ++i) bf[i] = (half_t)wrow[i];
      acc[0][an] = WMMA_F16(a0, bf, acc[0][an]);
      acc[1][an] = WMMA_F16(a1, bf, acc[1][an]);
    }
  }

  __syncthreads();
  {  // spill relu(h2 + b2) to LDS (f16)
    const int mrow = (lane >> 4) * 8;
    const int ncol = lane & 15;
#pragma unroll
    for (int am = 0; am < 2; ++am)
#pragma unroll
      for (int an = 0; an < 4; ++an) {
        const int n = wv * 64 + an * 16 + ncol;
        const float bn = f2_b2[n];
#pragma unroll
        for (int r = 0; r < 8; ++r) {
          const int m = am * 16 + mrow + r;
          Hs[m][n] = (half_t)fmaxf(acc[am][an][r] + bn, 0.f);
        }
      }
  }
  __syncthreads();

  if (tid < 96) {  // 32 rows x 3 output comps, each a K=512 dot
    const int row = tid / 3, cc = tid % 3;
    const float* w3 = f2_w3 + cc * 512;
    float s = 0.f;
    for (int k = 0; k < 512; ++k) s += (float)Hs[row][k] * w3[k];
    const int p = p0 + (row >> 4), g = row & 15;
    const float center = coarse[(size_t)b * 3072 + p * 3 + cc];
    const int pf = p * 16 + g;
    fine[((size_t)b * 16384 + pf) * 3 + cc] = s + f2_b3[cc] + center;
  }
}

// ---------------------------------------------------------------------------
extern "C" void kernel_launch(void* const* d_in, const int* in_sizes, int n_in,
                              void* d_out, int out_size, void* d_ws, size_t ws_size,
                              hipStream_t stream) {
  const float* x      = (const float*)d_in[0];
  const float* enc_w1 = (const float*)d_in[1];
  const float* enc_b1 = (const float*)d_in[2];
  const float* enc_w2 = (const float*)d_in[3];
  const float* enc_b2 = (const float*)d_in[4];
  const float* enc_w3 = (const float*)d_in[5];
  const float* enc_b3 = (const float*)d_in[6];
  const float* enc_w4 = (const float*)d_in[7];
  const float* enc_b4 = (const float*)d_in[8];
  const float* f1_w1  = (const float*)d_in[9];
  const float* f1_b1  = (const float*)d_in[10];
  const float* f1_w2  = (const float*)d_in[11];
  const float* f1_b2  = (const float*)d_in[12];
  const float* f1_w3  = (const float*)d_in[13];
  const float* f1_b3  = (const float*)d_in[14];
  const float* f2_w1  = (const float*)d_in[15];
  const float* f2_b1  = (const float*)d_in[16];
  const float* f2_w2  = (const float*)d_in[17];
  const float* f2_b2  = (const float*)d_in[18];
  const float* f2_w3  = (const float*)d_in[19];
  const float* f2_b3  = (const float*)d_in[20];

  float* out = (float*)d_out;  // [0:24576) coarse, [24576:) fine

  // workspace layout (floats); total ~160 MB
  float* ws = (float*)d_ws;
  size_t o = 0;
  float* enc1o = ws + o; o += (size_t)16384 * 128;
  float* enc2o = ws + o; o += (size_t)16384 * 256;
  float* gbuf  = ws + o; o += (size_t)8 * 256;
  float* cat   = ws + o; o += (size_t)16384 * 512;
  float* enc3o = ws + o; o += (size_t)16384 * 512;
  float* enc4o = ws + o; o += (size_t)16384 * 1024;
  float* feat  = ws + o; o += (size_t)8 * 1024;
  float* c1    = ws + o; o += (size_t)8 * 1024;
  float* c2    = ws + o; o += (size_t)8 * 1024;
  float* baseb = ws + o; o += (size_t)8 * 512;
  float* tg    = ws + o; o += (size_t)16 * 512;

  // encoder
  enc1_kernel<<<(16384 * 128) / 256, 256, 0, stream>>>(x, enc_w1, enc_b1, enc1o);
  gemm_wmma<false><<<dim3(256, 2), 256, 0, stream>>>(enc1o, enc_w2, enc_b2, enc2o,
                                                     16384, 256, 128);
  rowmax_kernel<<<(8 * 256 + 255) / 256, 256, 0, stream>>>(enc2o, gbuf, 256);
  concat_kernel<<<(16384 * 512) / 256, 256, 0, stream>>>(enc2o, gbuf, cat);
  gemm_wmma<true><<<dim3(256, 4), 256, 0, stream>>>(cat, enc_w3, enc_b3, enc3o,
                                                    16384, 512, 512);
  gemm_wmma<false><<<dim3(256, 8), 256, 0, stream>>>(enc3o, enc_w4, enc_b4, enc4o,
                                                     16384, 1024, 512);
  rowmax_kernel<<<(8 * 1024 + 255) / 256, 256, 0, stream>>>(enc4o, feat, 1024);

  // folding-1 (coarse) — M=8 padded into one 64-row tile, guarded
  gemm_wmma<true><<<dim3(1, 8), 256, 0, stream>>>(feat, f1_w1, f1_b1, c1, 8, 1024, 1024);
  gemm_wmma<true><<<dim3(1, 8), 256, 0, stream>>>(c1, f1_w2, f1_b2, c2, 8, 1024, 1024);
  gemm_wmma<false><<<dim3(1, 24), 256, 0, stream>>>(c2, f1_w3, f1_b3, out, 8, 3072, 1024);

  // folding-2 precompute + fused GEMM/tail
  base_kernel<<<(8 * 512 + 255) / 256, 256, 0, stream>>>(feat, f2_w1, f2_b1, baseb);
  tgrid_kernel<<<(16 * 512 + 255) / 256, 256, 0, stream>>>(f2_w1, tg);
  fold2_fused<<<4096, 256, 0, stream>>>(out, baseb, tg, f2_w1, f2_w2, f2_b2, f2_w3,
                                        f2_b3, out + 24576);
}
